// AdvOneLayer_88433376625403
// MI455X (gfx1250) — compile-verified, matrix operand order checked by reading, not run on hardware
//
#include <hip/hip_runtime.h>
#include <hip/hip_bf16.h>

typedef __attribute__((ext_vector_type(16))) _Float16 v16h;
typedef __attribute__((ext_vector_type(8)))  float    v8f;
typedef __attribute__((ext_vector_type(4)))  float    v4f;

#define BATCH   64
#define IN_DIM  1024
#define HID     4096
#define OUT_DIM 256
#define EPSILON 0.1f

// ---------------------------------------------------------------------------
// Partial row-sum of a 16-row tile over K in [kStart,kEnd) via WMMA:
// D = A * ones  =>  D[m,n] = sum_k A[m,k].  Because B is all-ones, the
// K-permutation inside the A operand is irrelevant: lane (0..15) supplies row
// m=lane, K 0..15 of each 32-chunk; lane (16..31) supplies row m=lane-16,
// K 16..31.  Each f32 input is split hi/lo into two f16 terms (after an exact
// power-of-2 scale) so the f32-accumulated result matches an f32 reduction to
// ~2^-22 relative error.
// Returns the 16 partial sums through lanes 0 (rows 0..7 in acc[0..7]) and
// lane 16 (rows 8..15).
// ---------------------------------------------------------------------------
template <bool DOABS>
__device__ inline v8f rowsum16_wmma_partial(const float* __restrict__ src,
                                            int ncols, int rowBase,
                                            int kStart, int kEnd, float scale) {
    const int lane = threadIdx.x & 31;
    const int m    = lane & 15;
    const int hi   = lane >> 4;

    const float* rowPtr = src + (size_t)(rowBase + m) * (size_t)ncols + hi * 16;

    v16h ones;
#pragma unroll
    for (int u = 0; u < 16; ++u) ones[u] = (_Float16)1.0f;

    v8f acc = {};

    for (int kb = kStart; kb < kEnd; kb += 32) {
        const v4f* p = (const v4f*)(rowPtr + kb);
        v4f q0 = p[0], q1 = p[1], q2 = p[2], q3 = p[3];

        float xs[16];
#pragma unroll
        for (int u = 0; u < 4; ++u) {
            xs[u]      = q0[u];
            xs[4 + u]  = q1[u];
            xs[8 + u]  = q2[u];
            xs[12 + u] = q3[u];
        }

        v16h ah, al;
#pragma unroll
        for (int u = 0; u < 16; ++u) {
            float v = xs[u];
            if (DOABS) v = fabsf(v);
            v *= scale;                       // exact (power of 2)
            _Float16 h = (_Float16)v;         // RNE
            float     r = v - (float)h;       // residual, |r| <= 2^-11 * |v|
            ah[u] = h;
            al[u] = (_Float16)r;
        }

        acc = __builtin_amdgcn_wmma_f32_16x16x32_f16(
            false, ah, false, ones, (short)0, acc, false, false);
        acc = __builtin_amdgcn_wmma_f32_16x16x32_f16(
            false, al, false, ones, (short)0, acc, false, false);
    }
    return acc;
}

// One block (128 threads = 4 waves) per 16-row tile; each wave reduces a
// quarter of the K range with WMMA, partials combined through LDS.
// Blocks 0..255:  w1norm[j] = ||W1[j,:]||_1   (4096 rows x 1024 cols, abs)
// Blocks 256..271: w2sum[i] = sum_j W2[i,j]   (256 rows x 4096 cols)
__global__ __launch_bounds__(128) void adv_reduce_kernel(
    const float* __restrict__ W1, const float* __restrict__ W2,
    float* __restrict__ w1norm, float* __restrict__ w2sum) {
    __shared__ float partial[4][16];

    const int b    = blockIdx.x;
    const int wave = threadIdx.x >> 5;
    const int lane = threadIdx.x & 31;

    const float* src;
    float*       dst;
    int ncols, rowBase;
    float scale, invScale;
    bool doAbs;
    if (b < HID / 16) {
        src = W1; dst = w1norm; ncols = IN_DIM; rowBase = b * 16;
        scale = 4096.0f; invScale = 1.0f / 4096.0f; doAbs = true;
    } else {
        src = W2; dst = w2sum; ncols = HID; rowBase = (b - HID / 16) * 16;
        scale = 256.0f; invScale = 1.0f / 256.0f; doAbs = false;
    }

    const int kQuarter = ncols >> 2;            // 256 (W1) or 1024 (W2)
    const int kStart   = wave * kQuarter;
    const int kEnd     = kStart + kQuarter;

    v8f acc;
    if (doAbs) acc = rowsum16_wmma_partial<true >(src, ncols, rowBase, kStart, kEnd, scale);
    else       acc = rowsum16_wmma_partial<false>(src, ncols, rowBase, kStart, kEnd, scale);

    // D layout: acc[r] -> row r on lanes 0..15, row 8+r on lanes 16..31 (all N
    // columns identical).  Lanes 0 and 16 of each wave deposit 8 sums each.
    if ((lane & 15) == 0) {
        const int rbase = (lane >> 4) * 8;
#pragma unroll
        for (int r = 0; r < 8; ++r) partial[wave][rbase + r] = acc[r];
    }
    __syncthreads();

    if (threadIdx.x < 16) {
        const float s = partial[0][threadIdx.x] + partial[1][threadIdx.x] +
                        partial[2][threadIdx.x] + partial[3][threadIdx.x];
        dst[rowBase + threadIdx.x] = s * invScale;
    }
}

// nn_output[b,i] = 4096 * rowsum(W2)[i] + bias2[i]   (independent of b)
__global__ __launch_bounds__(256) void adv_nnout_kernel(
    const float* __restrict__ w2sum, const float* __restrict__ bias2,
    float* __restrict__ out) {
    const int t = blockIdx.x * blockDim.x + threadIdx.x;  // 0 .. 16383
    const int i = t & (OUT_DIM - 1);
    out[t] = (float)HID * w2sum[i] + bias2[i];
}

__device__ inline float sgnf(float w) {
    return (w > 0.0f) ? 1.0f : ((w < 0.0f) ? -1.0f : 0.0f);
}

// pert[b,i,j] = -eps * y[b,i] * sign(W2[i,j]) * w1norm[j]
// 268 MB of pure streaming stores -> float4 non-temporal writes; W2 (4 MB)
// and w1norm (16 KB) stay hot in the 192 MB L2 across the 64 batch reuses.
__global__ __launch_bounds__(256) void adv_pert_kernel(
    const int* __restrict__ y, const float* __restrict__ W2,
    const float* __restrict__ w1norm, float* __restrict__ pert) {
    const size_t t   = (size_t)blockIdx.x * blockDim.x + threadIdx.x;
    const size_t e   = t * 4;                 // flat element index
    const int    j   = (int)(e & (HID - 1));
    const int    row = (int)(e >> 12);        // b*256 + i
    const int    i   = row & (OUT_DIM - 1);

    const float yv = (float)y[row];

    v4f o;
    if (yv != 0.0f) {
        const v4f w2 = *(const v4f*)(W2 + (size_t)i * HID + j);
        const v4f n  = *(const v4f*)(w1norm + j);
        const float c = -EPSILON * yv;
#pragma unroll
        for (int u = 0; u < 4; ++u) o[u] = c * sgnf(w2[u]) * n[u];
    } else {
        o = (v4f){0.0f, 0.0f, 0.0f, 0.0f};
    }
    __builtin_nontemporal_store(o, (v4f*)(pert + e));
}

extern "C" void kernel_launch(void* const* d_in, const int* in_sizes, int n_in,
                              void* d_out, int out_size, void* d_ws, size_t ws_size,
                              hipStream_t stream) {
    // setup_inputs order: x(0), y(1), W1(2), W2(3), bias1(4), bias2(5)
    const int*   y     = (const int*)d_in[1];
    const float* W1    = (const float*)d_in[2];
    const float* W2    = (const float*)d_in[3];
    const float* bias2 = (const float*)d_in[5];

    float* out    = (float*)d_out;
    float* nn_out = out;                           // [64, 256]
    float* pert   = out + (size_t)BATCH * OUT_DIM; // [64, 256, 4096]

    float* w1norm = (float*)d_ws;                  // 4096 floats
    float* w2sum  = w1norm + HID;                  // 256 floats

    // 1) WMMA row reductions (W1 L1-norms + W2 row sums):
    //    272 tiles x 4 waves, K split across waves, LDS combine.
    adv_reduce_kernel<<<HID / 16 + OUT_DIM / 16, 128, 0, stream>>>(W1, W2, w1norm, w2sum);

    // 2) nn_output broadcast: 16384 floats.
    adv_nnout_kernel<<<(BATCH * OUT_DIM) / 256, 256, 0, stream>>>(w2sum, bias2, nn_out);

    // 3) pert streaming write: 2^26 floats, 4 per thread.
    const size_t pert_elems = (size_t)BATCH * OUT_DIM * HID;   // 67,108,864
    adv_pert_kernel<<<(unsigned)(pert_elems / 4 / 256), 256, 0, stream>>>(y, W2, w1norm, pert);
}
